// Transformer_block_without_RoPE_11227044512283
// MI455X (gfx1250) — matrix operand order repaired and checked by reading, not compiled, not run on hardware
//
#include <hip/hip_runtime.h>
#include <cstdint>
#include <cstddef>

// ---------------------------------------------------------------------------
// CDNA5 (gfx1250) transformer block: bf16 WMMA GEMMs (TDM-fed, double
// buffered) + flash attention.
// ---------------------------------------------------------------------------

typedef __attribute__((ext_vector_type(16))) __bf16 v16bf;
typedef __attribute__((ext_vector_type(8)))  float  v8f;

#if defined(__has_builtin)
#if __has_builtin(__builtin_amdgcn_tensor_load_to_lds) && \
    __has_builtin(__builtin_amdgcn_s_wait_tensorcnt)
#define HAVE_TDM 1
#endif
#endif
#ifndef HAVE_TDM
#define HAVE_TDM 0
#endif

__device__ __forceinline__ v8f wmma_bf16f32(v16bf a, v16bf b, v8f c) {
  // D = A(16x32 bf16) * B(32x16 bf16) + C(16x16 f32)
  return __builtin_amdgcn_wmma_f32_16x16x32_bf16(false, a, false, b, (short)0, c,
                                                 false, false);
}

__device__ __forceinline__ v8f vzero8() {
  v8f z;
#pragma unroll
  for (int i = 0; i < 8; ++i) z[i] = 0.0f;
  return z;
}

// A-fragment (16 rows x 32 K), contiguous k order, same as frag_b:
// element j -> K = half*16 + j, row = lane&15.
__device__ __forceinline__ v16bf frag_a(const __bf16* base, int ld, int lane) {
  const int hf = (lane >> 4) & 1;
  const __bf16* p = base + (size_t)(lane & 15) * ld;
  v16bf r;
#pragma unroll
  for (int j = 0; j < 16; ++j) {
    const int k = (hf << 4) + j;
    r[j] = p[k];
  }
  return r;
}

// B-fragment (32 K x 16 N): element j -> K = half*16 + j, N = lane&15.
// base points at (N-row 0, K 0) of an N-major matrix (rows K-contiguous).
__device__ __forceinline__ v16bf frag_b(const __bf16* base, int ld, int lane) {
  const int hf = (lane >> 4) & 1;
  const __bf16* p = base + (size_t)(lane & 15) * ld + (hf << 4);
  v16bf r;
#pragma unroll
  for (int j = 0; j < 16; ++j) r[j] = p[j];
  return r;
}

#if HAVE_TDM
// ---------------------------------------------------------------------------
// Tensor Data Mover: 2-D tile load (bf16 elements) global -> LDS.
// D# packing per CDNA5 ISA ch.8 (group0: count/lds_addr/global_addr/type=2;
// group1: data_size=2B, tensor dims, tile dims, dim0 stride).
// ---------------------------------------------------------------------------
typedef unsigned int u32x4 __attribute__((ext_vector_type(4)));
typedef int i32x8 __attribute__((ext_vector_type(8)));
typedef int i32x4 __attribute__((ext_vector_type(4)));

__device__ __forceinline__ void tdm_load_tile_2d(unsigned lds_off,
                                                 const void* gptr,
                                                 unsigned tensor_d0,
                                                 unsigned tensor_d1,
                                                 unsigned stride_d0,
                                                 unsigned tile_d0,
                                                 unsigned tile_d1) {
  const unsigned long long ga = (unsigned long long)(size_t)gptr;
  u32x4 g0;
  g0[0] = 1u;       // count=1 (valid user descriptor), gather off
  g0[1] = lds_off;  // LDS byte address
  g0[2] = (unsigned)ga;
  g0[3] = ((unsigned)(ga >> 32) & 0x01FFFFFFu) | (2u << 30);  // addr hi | type=2
  i32x8 g1;
  g1[0] = (int)(1u << 16);  // workgroup_mask=0, data_size=1 (2 bytes)
  g1[1] = (int)((tensor_d0 & 0xFFFFu) << 16);                    // dim0 lo
  g1[2] = (int)((tensor_d0 >> 16) | ((tensor_d1 & 0xFFFFu) << 16));
  g1[3] = (int)((tensor_d1 >> 16) | ((tile_d0 & 0xFFFFu) << 16));
  g1[4] = (int)(tile_d1 & 0xFFFFu);  // tile_dim1 | tile_dim2=0
  g1[5] = (int)stride_d0;            // tensor_dim0_stride (elements)
  g1[6] = 0;
  g1[7] = 0;
  i32x4 z4 = {0, 0, 0, 0};
#if __clang_major__ >= 23
  i32x8 z8 = {0, 0, 0, 0, 0, 0, 0, 0};
  __builtin_amdgcn_tensor_load_to_lds(g0, g1, z4, z4, z8, 0);
#else
  __builtin_amdgcn_tensor_load_to_lds(g0, g1, z4, z4, 0);
#endif
}
#endif  // HAVE_TDM

// ---------------------------------------------------------------------------
// Elementwise f32 -> bf16 convert (weights).
// ---------------------------------------------------------------------------
__global__ void k_cvt_bf16(const float* __restrict__ s, __bf16* __restrict__ d,
                           size_t n) {
  for (size_t i = blockIdx.x * 256ull + threadIdx.x; i < n;
       i += (size_t)gridDim.x * 256ull)
    d[i] = (__bf16)s[i];
}

// ---------------------------------------------------------------------------
// RMSNorm: one block per token row, bf16 output.
// ---------------------------------------------------------------------------
__global__ __launch_bounds__(256) void k_rmsnorm(const float* __restrict__ x,
                                                 const float* __restrict__ g,
                                                 __bf16* __restrict__ o, int d) {
  __shared__ float red[256];
  const size_t row = blockIdx.x;
  const float* xr = x + row * (size_t)d;
  float ss = 0.f;
  for (int i = threadIdx.x; i < d; i += 256) {
    const float v = xr[i];
    ss += v * v;
  }
  red[threadIdx.x] = ss;
  __syncthreads();
  for (int s2 = 128; s2 > 0; s2 >>= 1) {
    if ((int)threadIdx.x < s2) red[threadIdx.x] += red[threadIdx.x + s2];
    __syncthreads();
  }
  const float rms = rsqrtf(red[0] / (float)d + 1e-5f);
  __bf16* orow = o + row * (size_t)d;
  for (int i = threadIdx.x; i < d; i += 256)
    orow[i] = (__bf16)(xr[i] * rms * g[i]);
}

// ---------------------------------------------------------------------------
// WMMA GEMM: C[M,N] = A[M,K] * B[N,K]^T  (both K-contiguous, bf16, f32 acc).
// Block tile 128x128, 8 waves of 64x32. K stepped by 32.
// Tiles are DMA'd by the Tensor Data Mover (issued by wave 0 behind a scalar
// branch; TDM ignores EXEC so an exec-masked issue would duplicate the DMA),
// double-buffered in LDS, synchronized with s_wait_tensorcnt + barrier.
// EPI: 0 = f32 store, 1 = bf16 store, 2 = +res f32 store,
//      3 = bf16( silu(gate) * C )  (SwiGLU fusion)
// ---------------------------------------------------------------------------
template <int EPI>
__global__ __launch_bounds__(256) void k_gemm(
    const __bf16* __restrict__ A, const __bf16* __restrict__ Bw,
    float* __restrict__ Cf, __bf16* __restrict__ Cb,
    const float* __restrict__ res, const float* __restrict__ gate, int M, int N,
    int K) {
  __shared__ __bf16 As[2][128][32];
  __shared__ __bf16 Bs[2][128][32];
  const int lane = threadIdx.x & 31;
  const int wv = threadIdx.x >> 5;  // 0..7
  const int hf = (lane >> 4) & 1;
  const int wm = (wv & 1) * 64;   // wave M offset in block tile
  const int wn = (wv >> 1) * 32;  // wave N offset in block tile
  const size_t m0 = (size_t)blockIdx.y * 128;
  const size_t n0 = (size_t)blockIdx.x * 128;

  v8f acc[4][2];
#pragma unroll
  for (int i = 0; i < 4; ++i)
#pragma unroll
    for (int j = 0; j < 2; ++j) acc[i][j] = vzero8();

  const int nk = K / 32;

#if HAVE_TDM
  // Uniform wave id -> scalar branch around the TDM issue.
  const int wvu = __builtin_amdgcn_readfirstlane(threadIdx.x >> 5);
  if (wvu == 0) {
    tdm_load_tile_2d((unsigned)(size_t)&As[0][0][0], A + m0 * K,
                     (unsigned)K, (unsigned)M, (unsigned)K, 32u, 128u);
    tdm_load_tile_2d((unsigned)(size_t)&Bs[0][0][0], Bw + n0 * K,
                     (unsigned)K, (unsigned)N, (unsigned)K, 32u, 128u);
    __builtin_amdgcn_s_wait_tensorcnt(0);
  }
  __syncthreads();  // buffer 0 visible to all waves
#endif

  for (int kb = 0; kb < nk; ++kb) {
#if HAVE_TDM
    const int cur = kb & 1;
    if (wvu == 0 && kb + 1 < nk) {
      const size_t k1 = (size_t)(kb + 1) * 32;
      tdm_load_tile_2d((unsigned)(size_t)&As[cur ^ 1][0][0], A + m0 * K + k1,
                       (unsigned)K, (unsigned)M, (unsigned)K, 32u, 128u);
      tdm_load_tile_2d((unsigned)(size_t)&Bs[cur ^ 1][0][0], Bw + n0 * K + k1,
                       (unsigned)K, (unsigned)N, (unsigned)K, 32u, 128u);
    }
#else
    const int cur = 0;
    const int lr = threadIdx.x >> 2;        // 0..63
    const int lc = (threadIdx.x & 3) << 3;  // 0,8,16,24 (16B)
    const int k0 = kb * 32;
#pragma unroll
    for (int p = 0; p < 2; ++p) {
      const int row = lr + p * 64;
      *(uint4*)(&As[0][row][lc]) =
          *(const uint4*)(A + (m0 + row) * (size_t)K + k0 + lc);
      *(uint4*)(&Bs[0][row][lc]) =
          *(const uint4*)(Bw + (n0 + row) * (size_t)K + k0 + lc);
    }
    __syncthreads();
#endif

    v16bf af[4];
    v16bf bfg[2];
#pragma unroll
    for (int mt = 0; mt < 4; ++mt)
      af[mt] = frag_a(&As[cur][wm + mt * 16][0], 32, lane);
#pragma unroll
    for (int nt = 0; nt < 2; ++nt)
      bfg[nt] = frag_b(&Bs[cur][wn + nt * 16][0], 32, lane);
#pragma unroll
    for (int mt = 0; mt < 4; ++mt)
#pragma unroll
      for (int nt = 0; nt < 2; ++nt)
        acc[mt][nt] = wmma_bf16f32(af[mt], bfg[nt], acc[mt][nt]);

#if HAVE_TDM
    if (kb + 1 < nk) {
      if (wvu == 0) __builtin_amdgcn_s_wait_tensorcnt(0);
      __syncthreads();  // next buffer ready; this buffer free to overwrite
    }
#else
    __syncthreads();
#endif
  }

  // Epilogue: C element v -> row M = v + 8*half, col N = lane&15.
#pragma unroll
  for (int mt = 0; mt < 4; ++mt)
#pragma unroll
    for (int nt = 0; nt < 2; ++nt)
#pragma unroll
      for (int v = 0; v < 8; ++v) {
        const size_t r = m0 + wm + mt * 16 + v + 8 * hf;
        const size_t c = n0 + wn + nt * 16 + (lane & 15);
        const size_t idx = r * (size_t)N + c;
        const float cv = acc[mt][nt][v];
        if (EPI == 0) {
          Cf[idx] = cv;
        } else if (EPI == 1) {
          Cb[idx] = (__bf16)cv;
        } else if (EPI == 2) {
          Cf[idx] = cv + res[idx];
        } else {
          const float gt = gate[idx];
          const float si = gt / (1.0f + __expf(-gt));
          Cb[idx] = (__bf16)(si * cv);
        }
      }
}

// ---------------------------------------------------------------------------
// Flash attention (causal), bf16 WMMA, f32 online softmax.
// Grid: (S/64, B*H). Block: 128 threads = 4 waves; each wave owns a 16-query
// tile, streams keys 32 at a time. Q/K/V/O layout: [B, S, H*64] (token-major)
// so the surrounding GEMMs consume/produce them directly.
// ---------------------------------------------------------------------------
__global__ __launch_bounds__(128) void k_attn(const __bf16* __restrict__ Q,
                                              const __bf16* __restrict__ Kc,
                                              const __bf16* __restrict__ V,
                                              __bf16* __restrict__ O) {
  constexpr int S = 2048, D = 1024, DK = 64, H = 16;
  __shared__ __bf16 p_lds[4][16][32];  // per-wave P tile (C-layout -> A-layout)
  const int lane = threadIdx.x & 31;
  const int wv = threadIdx.x >> 5;
  const int hf = (lane >> 4) & 1;
  const int nl = lane & 15;
  const int bh = blockIdx.y;
  const int b = bh / H;
  const int h = bh % H;
  const int q0 = blockIdx.x * 64 + wv * 16;

  const size_t baseQ = ((size_t)b * S + q0) * D + (size_t)h * DK;
  const v16bf qa0 = frag_a(Q + baseQ, D, lane);       // d 0..31
  const v16bf qa1 = frag_a(Q + baseQ + 32, D, lane);  // d 32..63

  v8f o[4];
  float mrun[8], lrun[8];
#pragma unroll
  for (int nt = 0; nt < 4; ++nt) o[nt] = vzero8();
#pragma unroll
  for (int v = 0; v < 8; ++v) {
    mrun[v] = -1e30f;
    lrun[v] = 0.0f;
  }

  const int qhi = q0 + 15;
  for (int j0 = 0; j0 <= qhi; j0 += 32) {
    // --- scores S = Q K^T (two 16-key subtiles, dk=64 => 2 WMMAs each) ---
    v8f s[2];
#pragma unroll
    for (int sub = 0; sub < 2; ++sub) {
      s[sub] = vzero8();
      const size_t kb = ((size_t)b * S + j0 + sub * 16) * D + (size_t)h * DK;
      const v16bf kb0 = frag_b(Kc + kb, D, lane);       // contraction d 0..31
      const v16bf kb1 = frag_b(Kc + kb + 32, D, lane);  // contraction d 32..63
      s[sub] = wmma_bf16f32(qa0, kb0, s[sub]);
      s[sub] = wmma_bf16f32(qa1, kb1, s[sub]);
    }
    // --- causal mask + online softmax (row = v + 8*half, 16 lanes/row) ---
#pragma unroll
    for (int v = 0; v < 8; ++v) {
      const int qm = q0 + v + 8 * hf;
      float s0 = s[0][v] * 0.125f;  // 1/sqrt(64)
      float s1 = s[1][v] * 0.125f;
      if (j0 + nl > qm) s0 = -1e30f;
      if (j0 + 16 + nl > qm) s1 = -1e30f;
      float mx = fmaxf(s0, s1);
#pragma unroll
      for (int d2 = 1; d2 < 16; d2 <<= 1) mx = fmaxf(mx, __shfl_xor(mx, d2, 32));
      const float mnew = fmaxf(mrun[v], mx);
      const float alpha = __expf(mrun[v] - mnew);
      const float e0 = __expf(s0 - mnew);
      const float e1 = __expf(s1 - mnew);
      float rs = e0 + e1;
#pragma unroll
      for (int d2 = 1; d2 < 16; d2 <<= 1) rs += __shfl_xor(rs, d2, 32);
      lrun[v] = lrun[v] * alpha + rs;
      mrun[v] = mnew;
      p_lds[wv][v + 8 * hf][nl] = (__bf16)e0;
      p_lds[wv][v + 8 * hf][16 + nl] = (__bf16)e1;
#pragma unroll
      for (int nt = 0; nt < 4; ++nt) o[nt][v] *= alpha;
    }
    // Same-wave LDS RAW: DS ops are in-order, fence the counter explicitly.
    asm volatile("s_wait_dscnt 0" ::: "memory");
    const v16bf pa = frag_a(&p_lds[wv][0][0], 32, lane);  // P as A (16x32)
    // --- O += P V : 4 dk-subtiles of 16 ---
#pragma unroll
    for (int nt = 0; nt < 4; ++nt) {
      v16bf vb;
#pragma unroll
      for (int j = 0; j < 16; ++j) {
        const int key = j0 + hf * 16 + j;  // contraction = key index
        vb[j] = V[((size_t)b * S + key) * D + (size_t)h * DK + nt * 16 + nl];
      }
      o[nt] = wmma_bf16f32(pa, vb, o[nt]);
    }
  }

  // --- normalize + store bf16 into [B,S,H*64] ---
#pragma unroll
  for (int nt = 0; nt < 4; ++nt)
#pragma unroll
    for (int v = 0; v < 8; ++v) {
      const float val = o[nt][v] / lrun[v];
      O[((size_t)b * S + q0 + v + 8 * hf) * D + (size_t)h * DK + nt * 16 + nl] =
          (__bf16)val;
    }
}

// ---------------------------------------------------------------------------
// Host orchestration.
// ---------------------------------------------------------------------------
extern "C" void kernel_launch(void* const* d_in, const int* in_sizes, int n_in,
                              void* d_out, int out_size, void* d_ws,
                              size_t ws_size, hipStream_t stream) {
  (void)in_sizes; (void)n_in; (void)out_size; (void)ws_size;
  constexpr int B = 4, S = 2048, D = 1024, H = 16, FF = 4096;
  constexpr size_t M = (size_t)B * S;  // 8192 tokens

  const float* x  = (const float*)d_in[0];
  const float* Wq = (const float*)d_in[1];
  const float* Wk = (const float*)d_in[2];
  const float* Wv = (const float*)d_in[3];
  const float* Wo = (const float*)d_in[4];
  const float* g1 = (const float*)d_in[5];
  const float* g2 = (const float*)d_in[6];
  const float* W1 = (const float*)d_in[7];
  const float* W3 = (const float*)d_in[8];
  const float* W2 = (const float*)d_in[9];
  float* out = (float*)d_out;

  char* ws = (char*)d_ws;
  size_t off = 0;
  auto take = [&](size_t bytes) -> void* {
    void* p = ws + off;
    off += (bytes + 255) & ~(size_t)255;
    return p;
  };

  __bf16* h1  = (__bf16*)take(M * D * 2);
  __bf16* wqb = (__bf16*)take((size_t)D * D * 2);
  __bf16* wkb = (__bf16*)take((size_t)D * D * 2);
  __bf16* wvb = (__bf16*)take((size_t)D * D * 2);
  __bf16* wob = (__bf16*)take((size_t)D * D * 2);
  __bf16* w1b = (__bf16*)take((size_t)FF * D * 2);
  __bf16* w3b = (__bf16*)take((size_t)FF * D * 2);
  __bf16* w2b = (__bf16*)take((size_t)D * FF * 2);
  __bf16* Qb  = (__bf16*)take(M * D * 2);
  __bf16* Kb  = (__bf16*)take(M * D * 2);
  __bf16* Vb  = (__bf16*)take(M * D * 2);
  __bf16* at  = (__bf16*)take(M * D * 2);
  float*  x2  = (float*)take(M * D * 4);
  __bf16* h2  = (__bf16*)take(M * D * 2);
  float*  gatef = (float*)take(M * FF * 4);
  __bf16* ffb = (__bf16*)take(M * FF * 2);

  auto cvt = [&](const float* s, __bf16* d, size_t n) {
    k_cvt_bf16<<<dim3(2048), dim3(256), 0, stream>>>(s, d, n);
  };
  cvt(Wq, wqb, (size_t)D * D);
  cvt(Wk, wkb, (size_t)D * D);
  cvt(Wv, wvb, (size_t)D * D);
  cvt(Wo, wob, (size_t)D * D);
  cvt(W1, w1b, (size_t)FF * D);
  cvt(W3, w3b, (size_t)FF * D);
  cvt(W2, w2b, (size_t)D * FF);

  const dim3 blk(256);
  // h1 = rmsnorm(x, g1)
  k_rmsnorm<<<dim3((unsigned)M), blk, 0, stream>>>(x, g1, h1, D);
  // Q/K/V projections (bf16 out)
  k_gemm<1><<<dim3(D / 128, M / 128), blk, 0, stream>>>(h1, wqb, nullptr, Qb,
                                                        nullptr, nullptr, (int)M, D, D);
  k_gemm<1><<<dim3(D / 128, M / 128), blk, 0, stream>>>(h1, wkb, nullptr, Kb,
                                                        nullptr, nullptr, (int)M, D, D);
  k_gemm<1><<<dim3(D / 128, M / 128), blk, 0, stream>>>(h1, wvb, nullptr, Vb,
                                                        nullptr, nullptr, (int)M, D, D);
  // flash attention -> attn (bf16, [B,S,D])
  k_attn<<<dim3(S / 64, B * H), dim3(128), 0, stream>>>(Qb, Kb, Vb, at);
  // x2 = attn @ Wo^T + x   (f32)
  k_gemm<2><<<dim3(D / 128, M / 128), blk, 0, stream>>>(at, wob, x2, nullptr, x,
                                                        nullptr, (int)M, D, D);
  // h2 = rmsnorm(x2, g2)
  k_rmsnorm<<<dim3((unsigned)M), blk, 0, stream>>>(x2, g2, h2, D);
  // gate = h2 @ W1^T (f32)
  k_gemm<0><<<dim3(FF / 128, M / 128), blk, 0, stream>>>(h2, w1b, gatef, nullptr,
                                                         nullptr, nullptr, (int)M, FF, D);
  // ff = bf16( silu(gate) * (h2 @ W3^T) )
  k_gemm<3><<<dim3(FF / 128, M / 128), blk, 0, stream>>>(h2, w3b, nullptr, ffb,
                                                         nullptr, gatef, (int)M, FF, D);
  // out = ff @ W2^T + x2  (f32)
  k_gemm<2><<<dim3(D / 128, M / 128), blk, 0, stream>>>(ffb, w2b, out, nullptr,
                                                        x2, nullptr, (int)M, D, FF);
}
